// LoRALayer_40372692583098
// MI455X (gfx1250) — matrix operand order repaired and checked by reading, not compiled, not run on hardware
//
#include <hip/hip_runtime.h>
#include <hip/hip_bf16.h>

// Problem dims (fixed by the reference)
#define OUT_DIM 4096
#define IN_DIM  4096
#define RANK    16
#define NTOK    8192

typedef __bf16 v16bf __attribute__((ext_vector_type(16)));
typedef __bf16 v8bf  __attribute__((ext_vector_type(8)));
typedef float  v8f   __attribute__((ext_vector_type(8)));

// ---------------------------------------------------------------------------
// Kernel 1: Wb[o][i] = bf16( W[o][i] + sum_r A[o][r] * B[r][i] )
// ---------------------------------------------------------------------------
__global__ __launch_bounds__(256) void fuse_w_kernel(
    const float* __restrict__ W, const float* __restrict__ A,
    const float* __restrict__ B, __bf16* __restrict__ Wb)
{
    const size_t idx = (size_t)blockIdx.x * 256 + threadIdx.x;  // over OUT*IN
    const int o = (int)(idx / IN_DIM);
    const int i = (int)(idx % IN_DIM);
    float acc = W[idx];
#pragma unroll
    for (int r = 0; r < RANK; ++r)
        acc = fmaf(A[o * RANK + r], B[(size_t)r * IN_DIM + i], acc);
    Wb[idx] = (__bf16)acc;
}

// ---------------------------------------------------------------------------
// Kernel 2: xT[t][i] = bf16( x[i][t] )   (x is [IN][NTOK] row-major)
// LDS tile transpose so both global read and write are coalesced.
// ---------------------------------------------------------------------------
__global__ __launch_bounds__(256) void transpose_cvt_kernel(
    const float* __restrict__ x, __bf16* __restrict__ xT)
{
    __shared__ float tile[32][33];
    const int i0 = blockIdx.x * 32;   // IN dim
    const int t0 = blockIdx.y * 32;   // NTOK dim
    const int tx = threadIdx.x;       // 0..31
    const int ty = threadIdx.y;       // 0..7
#pragma unroll
    for (int j = 0; j < 32; j += 8)
        tile[ty + j][tx] = x[(size_t)(i0 + ty + j) * NTOK + (t0 + tx)];
    __syncthreads();
#pragma unroll
    for (int j = 0; j < 32; j += 8)
        xT[(size_t)(t0 + ty + j) * IN_DIM + (i0 + tx)] = (__bf16)tile[tx][ty + j];
}

// ---------------------------------------------------------------------------
// Kernel 3: out = Wb @ x via v_wmma_f32_16x16x32_bf16.
// Block = 256 threads = 8 waves arranged 2(M) x 4(N); each wave computes a
// 64x64 output tile (4 A-frags x 4 B-frags = 16 WMMA / K-step) => block tile
// 128(M) x 256(N). Grid = (OUT/128, NTOK/256) = (32, 32).
//
// K loop is unrolled by 64 with ping/pong fragment buffers so every fragment
// load has a full 16-WMMA window in flight before its s_wait_loadcnt.
//
// Fragment layouts (CDNA5 wave32, 16-bit):
//   A 16x32 : row = lane&15 ; hi=(lane>>4)*8 ; elems 0..7 -> K=2*hi..2*hi+7,
//             elems 8..15 -> K=2*hi+8..2*hi+15  (two 16B loads, row-major Wb)
//   B 32x16 : col = lane&15 ; K = (lane>>4)*16 + 0..15 contiguous
//             (one 32B load from xT, which stores K contiguously per token)
//   C/D     : VGPR r -> M = r + (lane>>4)*8, N = lane&15
// ---------------------------------------------------------------------------
__device__ __forceinline__ v16bf load_afrag(const __bf16* row, int kk, int hi)
{
    v8bf alo = *(const v8bf*)(row + kk + 2 * hi);
    v8bf ahi = *(const v8bf*)(row + kk + 2 * hi + 8);
    return __builtin_shufflevector(alo, ahi,
        0, 1, 2, 3, 4, 5, 6, 7, 8, 9, 10, 11, 12, 13, 14, 15);
}

__global__ __launch_bounds__(256) void lora_gemm_kernel(
    const __bf16* __restrict__ Wb, const __bf16* __restrict__ xT,
    float* __restrict__ out)
{
    const int lane = threadIdx.x & 31;
    const int wave = threadIdx.x >> 5;
    const int wm = wave & 1;        // 0..1
    const int wn = wave >> 1;       // 0..3

    const int M0 = blockIdx.x * 128 + wm * 64;   // wave's M base (64 rows)
    const int N0 = blockIdx.y * 256 + wn * 64;   // wave's N base (64 cols)

    const int l16  = lane & 15;
    const int hi   = (lane >> 4) * 8;    // A K sub-offset
    const int kb16 = (lane >> 4) * 16;   // B K sub-offset

    const __bf16* arow[4];
#pragma unroll
    for (int mt = 0; mt < 4; ++mt)
        arow[mt] = Wb + (size_t)(M0 + mt * 16 + l16) * IN_DIM;

    const __bf16* bcol[4];
#pragma unroll
    for (int nt = 0; nt < 4; ++nt)
        bcol[nt] = xT + (size_t)(N0 + nt * 16 + l16) * IN_DIM;

    v8f acc[4][4];
#pragma unroll
    for (int mt = 0; mt < 4; ++mt)
#pragma unroll
        for (int nt = 0; nt < 4; ++nt)
            acc[mt][nt] = (v8f)(0.0f);

    // ---- Software pipeline: ping (stage 0) / pong (stage 1) fragments ----
    v16bf a0[4], b0[4], a1[4], b1[4];

    // Prologue: stage-0 fragments for kk = 0
#pragma unroll
    for (int mt = 0; mt < 4; ++mt) a0[mt] = load_afrag(arow[mt], 0, hi);
#pragma unroll
    for (int nt = 0; nt < 4; ++nt) b0[nt] = *(const v16bf*)(bcol[nt] + kb16);

    for (int kk = 0; kk < IN_DIM; kk += 64) {
        // Issue stage-1 loads (kk+32) before consuming stage-0.
#pragma unroll
        for (int mt = 0; mt < 4; ++mt) a1[mt] = load_afrag(arow[mt], kk + 32, hi);
#pragma unroll
        for (int nt = 0; nt < 4; ++nt) b1[nt] = *(const v16bf*)(bcol[nt] + kk + 32 + kb16);

        // 16 WMMAs on stage-0 (covers stage-1 load latency).
#pragma unroll
        for (int mt = 0; mt < 4; ++mt)
#pragma unroll
            for (int nt = 0; nt < 4; ++nt)
                acc[mt][nt] = __builtin_amdgcn_wmma_f32_16x16x32_bf16(
                    false, a0[mt], false, b0[nt],
                    (short)0, acc[mt][nt], false, false);

        // Issue stage-0 loads for next iteration (kk+64), guarded at the tail.
        if (kk + 64 < IN_DIM) {
#pragma unroll
            for (int mt = 0; mt < 4; ++mt) a0[mt] = load_afrag(arow[mt], kk + 64, hi);
#pragma unroll
            for (int nt = 0; nt < 4; ++nt) b0[nt] = *(const v16bf*)(bcol[nt] + kk + 64 + kb16);
        }

        // 16 WMMAs on stage-1 (covers next stage-0 load latency).
#pragma unroll
        for (int mt = 0; mt < 4; ++mt)
#pragma unroll
            for (int nt = 0; nt < 4; ++nt)
                acc[mt][nt] = __builtin_amdgcn_wmma_f32_16x16x32_bf16(
                    false, a1[mt], false, b1[nt],
                    (short)0, acc[mt][nt], false, false);
    }

    // --- Store C/D: VGPR r -> M = r + (lane>>4)*8, N = lane&15 ---
    const int mrow_off = (lane >> 4) * 8;
#pragma unroll
    for (int mt = 0; mt < 4; ++mt) {
#pragma unroll
        for (int nt = 0; nt < 4; ++nt) {
#pragma unroll
            for (int r = 0; r < 8; ++r) {
                const int m = M0 + mt * 16 + mrow_off + r;
                const int n = N0 + nt * 16 + l16;
                out[(size_t)m * NTOK + n] = acc[mt][nt][r];
            }
        }
    }
}

// ---------------------------------------------------------------------------
// Launch: inputs in setup_inputs() order: x, weight, A, B (all fp32).
// Workspace layout: [ Wb : OUT*IN bf16 = 32MB ][ xT : NTOK*IN bf16 = 64MB ]
// ---------------------------------------------------------------------------
extern "C" void kernel_launch(void* const* d_in, const int* in_sizes, int n_in,
                              void* d_out, int out_size, void* d_ws, size_t ws_size,
                              hipStream_t stream) {
    const float* x  = (const float*)d_in[0];   // [IN, NTOK]
    const float* W  = (const float*)d_in[1];   // [OUT, IN]
    const float* A  = (const float*)d_in[2];   // [OUT, RANK]
    const float* B  = (const float*)d_in[3];   // [RANK, IN]
    float* out = (float*)d_out;                // [OUT, NTOK]

    __bf16* Wb = (__bf16*)d_ws;
    __bf16* xT = (__bf16*)((char*)d_ws + (size_t)OUT_DIM * IN_DIM * sizeof(__bf16));

    // 1) Wb = bf16(W + A@B)
    {
        const size_t nElem = (size_t)OUT_DIM * IN_DIM;
        fuse_w_kernel<<<dim3((unsigned)(nElem / 256)), dim3(256), 0, stream>>>(W, A, B, Wb);
    }
    // 2) xT = bf16(x)^T  -> [NTOK][IN]
    {
        dim3 grid(IN_DIM / 32, NTOK / 32);
        dim3 block(32, 8);
        transpose_cvt_kernel<<<grid, block, 0, stream>>>(x, xT);
    }
    // 3) out = Wb @ x  (WMMA bf16, fp32 accumulate)
    {
        dim3 grid(OUT_DIM / 128, NTOK / 256);
        lora_gemm_kernel<<<grid, dim3(256), 0, stream>>>(Wb, xT, out);
    }
}
